// Discriminator_21818433863763
// MI455X (gfx1250) — matrix operand order repaired and checked by reading, not compiled, not run on hardware
//
#include <hip/hip_runtime.h>
#include <hip/hip_bf16.h>

// LSTM: B=128, T=1024, D=128, H=512. Recurrence fused as
//   g = [x_t | h_{t-1}] @ [W_ih | W_hh]^T + (b_ih+b_hh),  K = 640
// computed with v_wmma_f32_16x16x32_bf16; weights held resident in VGPRs
// (20 x v16bf = 160 VGPRs/wave) for the whole 1024-step scan.
// A fetch is a 3-deep software pipeline pinned with sched_group_barrier
// (0x8 = MFMA/WMMA group, 0x20 = VMEM-read group) so loads overlap WMMAs.

typedef __attribute__((ext_vector_type(16))) __bf16 v16bf;
typedef __attribute__((ext_vector_type(8)))  float  v8f;

#define LSTM_B   128
#define LSTM_T   1024
#define LSTM_D   128
#define LSTM_H   512
#define LSTM_G   2048   // 4H
#define LSTM_K   640    // D + H
#define NBLOCKS  256    // 8 batch-tiles * 32 h-slices

// ---- workspace layout (bytes) ----
#define O_BAR    0                         // 2 ints (barrier cnt/gen)
#define O_BSUM   256                       // 2048 f32
#define O_HLAST  8448                      // 128*512 f32
#define O_HB0    270592                    // 128*512 bf16 (ping)
#define O_HB1    401664                    // 128*512 bf16 (pong)
#define O_WC     532736                    // 2048*640 bf16
#define O_XB     3154176                   // 128*1024*128 bf16
// total ~36.8 MB

__global__ void lstm_prep(const float* __restrict__ x,
                          const float* __restrict__ Wih,
                          const float* __restrict__ Whh,
                          const float* __restrict__ bih,
                          const float* __restrict__ bhh,
                          char* __restrict__ ws)
{
  int*    bar  = (int*)   (ws + O_BAR);
  float*  bsum = (float*) (ws + O_BSUM);
  __bf16* hb0  = (__bf16*)(ws + O_HB0);
  __bf16* hb1  = (__bf16*)(ws + O_HB1);
  __bf16* Wc   = (__bf16*)(ws + O_WC);
  __bf16* xb   = (__bf16*)(ws + O_XB);

  long tid = (long)blockIdx.x * blockDim.x + threadIdx.x;
  long nth = (long)gridDim.x * blockDim.x;

  if (tid < 2) bar[tid] = 0;

  for (long i = tid; i < (long)LSTM_G * LSTM_K; i += nth) {
    int n = (int)(i / LSTM_K), k = (int)(i % LSTM_K);
    float w = (k < LSTM_D) ? Wih[(long)n * LSTM_D + k]
                           : Whh[(long)n * LSTM_H + (k - LSTM_D)];
    Wc[i] = (__bf16)w;
  }
  for (long i = tid; i < LSTM_G; i += nth) bsum[i] = bih[i] + bhh[i];
  for (long i = tid; i < (long)LSTM_B * LSTM_T * LSTM_D; i += nth)
    xb[i] = (__bf16)x[i];
  for (long i = tid; i < (long)LSTM_B * LSTM_H; i += nth) {
    hb0[i] = (__bf16)0.f; hb1[i] = (__bf16)0.f;
  }
}

__launch_bounds__(128, 1)
__global__ void lstm_scan(char* __restrict__ ws)
{
  const float*  bsum  = (const float*) (ws + O_BSUM);
  float*        hlast = (float*)       (ws + O_HLAST);
  __bf16*       hb0   = (__bf16*)      (ws + O_HB0);
  __bf16*       hb1   = (__bf16*)      (ws + O_HB1);
  const __bf16* Wc    = (const __bf16*)(ws + O_WC);
  const __bf16* xb    = (const __bf16*)(ws + O_XB);
  int*          bar   = (int*)         (ws + O_BAR);

  __shared__ float lds_g[4][8][32];   // 4 gate tiles in C layout
  __shared__ float lds_c[256];        // cell state, lives here all 1024 steps

  const int mtile  = blockIdx.x & 7;       // batch tile (16 rows)
  const int hslice = blockIdx.x >> 3;      // hidden slice (16 cols)
  const int gi     = threadIdx.x >> 5;     // wave id == gate id (i,f,g,o)
  const int lane   = threadIdx.x & 31;
  const int m0     = mtile * 16;
  const int h0     = hslice * 16;
  const int n0     = gi * LSTM_H + h0;     // column in 4H gate space

  lds_c[threadIdx.x]       = 0.f;
  lds_c[threadIdx.x + 128] = 0.f;
  __syncthreads();

  union U32 { v16bf v; uint4 u[2]; };

  // ---- preload B fragments: resident in VGPRs for the whole scan ----
  // B 32x16 bf16 layout: lane = N (mod 16), lanes>=16 hold K+16; 16 contiguous K per lane.
  const int nlane = n0 + (lane & 15);
  const int kbB   = (lane >> 4) * 16;
  v16bf breg[20];
#pragma unroll
  for (int ks = 0; ks < 20; ++ks) {
    const __bf16* p = Wc + (long)nlane * LSTM_K + ks * 32 + kbB;
    U32 w;
    w.u[0] = *(const uint4*)(p);
    w.u[1] = *(const uint4*)(p + 8);
    breg[ks] = w.v;
  }
  const float bias = bsum[nlane];

  // A 16x32 bf16 layout: lane = M (mod 16); lanes<16 hold K 0-15, lanes>=16 K 16-31 (contiguous, matches B).
  const int mA  = m0 + (lane & 15);
  const int kbA = (lane >> 4) * 16;
  const __bf16* xrow = xb + (long)mA * LSTM_T * LSTM_D + kbA;

  for (int t = 0; t < LSTM_T; ++t) {
    const __bf16* hread  = (t & 1) ? hb1 : hb0;
    __bf16*       hwrite = (t & 1) ? hb0 : hb1;

    v8f acc;
#pragma unroll
    for (int v = 0; v < 8; ++v) acc[v] = bias;

    const __bf16* xr = xrow + (long)t * LSTM_D;
    const __bf16* hr = hread + (long)mA * LSTM_H + kbA;

    // branchless prefetch of next timestep's x row (keeps one sched region)
    __builtin_prefetch(xr + ((t + 1 < LSTM_T) ? LSTM_D : 0), 0, 3);

    // k-step ks: A fragment pointer (constant-folds under full unroll)
    auto aptr = [&](int ks) -> const uint4* {
      return (ks < 4) ? (const uint4*)(xr + ks * 32)
                      : (const uint4*)(hr + (ks - 4) * 32);
    };

    // ---- 3-deep pipelined A fetch (rotating buffer, ks % 3) ----
    U32 abuf[3];
#pragma unroll
    for (int j = 0; j < 3; ++j) {
      abuf[j].u[0] = aptr(j)[0];
      abuf[j].u[1] = aptr(j)[1];
    }
#pragma unroll
    for (int ks = 0; ks < 20; ++ks) {
      acc = __builtin_amdgcn_wmma_f32_16x16x32_bf16(
          false, abuf[ks % 3].v, false, breg[ks], (short)0, acc, false, false);
      if (ks + 3 < 20) {
        abuf[ks % 3].u[0] = aptr(ks + 3)[0];
        abuf[ks % 3].u[1] = aptr(ks + 3)[1];
      }
    }
    // Pin the pipeline: 6 loads up front, then 17x [1 WMMA, 2 loads], 3 WMMAs.
    __builtin_amdgcn_sched_group_barrier(0x020, 6, 0);   // VMEM read x6
#pragma unroll
    for (int ks = 0; ks < 17; ++ks) {
      __builtin_amdgcn_sched_group_barrier(0x008, 1, 0); // WMMA x1
      __builtin_amdgcn_sched_group_barrier(0x020, 2, 0); // VMEM read x2
    }
    __builtin_amdgcn_sched_group_barrier(0x008, 3, 0);   // WMMA x3

    // exchange gate tiles through LDS (C layout: VGPR v = M row, lane = N)
#pragma unroll
    for (int v = 0; v < 8; ++v) lds_g[gi][v][lane] = acc[v];
    __syncthreads();

    // elementwise LSTM cell update: 256 elements, 2 per thread
#pragma unroll
    for (int r = 0; r < 2; ++r) {
      int e  = threadIdx.x + r * 128;
      int ml = e >> 4, nl = e & 15;
      int ln = ((ml & 8) ? 16 : 0) + nl;
      int vv = ml & 7;
      float g_i = lds_g[0][vv][ln];
      float g_f = lds_g[1][vv][ln];
      float g_g = lds_g[2][vv][ln];
      float g_o = lds_g[3][vv][ln];
      float si = 1.f / (1.f + __expf(-g_i));
      float sf = 1.f / (1.f + __expf(-g_f));
      float tg = tanhf(g_g);
      float so = 1.f / (1.f + __expf(-g_o));
      float c  = sf * lds_c[e] + si * tg;
      lds_c[e] = c;
      float h  = so * tanhf(c);
      long  gx = (long)(m0 + ml) * LSTM_H + (h0 + nl);
      hwrite[gx] = (__bf16)h;
      if (t == LSTM_T - 1) hlast[gx] = h;
    }

    // grid-wide sense-reversing barrier (one per timestep)
    __threadfence();
    __syncthreads();
    if (threadIdx.x == 0) {
      volatile int* vgen = bar + 1;
      int g = *vgen;
      if (atomicAdd(bar, 1) == NBLOCKS - 1) {
        atomicExch(bar, 0);
        __threadfence();
        atomicAdd((int*)(bar + 1), 1);
      } else {
        while (*vgen == g) __builtin_amdgcn_s_sleep(2);
      }
    }
    __syncthreads();
  }
}

__global__ void lstm_head(const char* __restrict__ ws,
                          const float* __restrict__ Wfc,
                          const float* __restrict__ bfc,
                          float* __restrict__ out)
{
  const float* hlast = (const float*)(ws + O_HLAST);
  __shared__ float red[64];
  int b = blockIdx.x;
  float s = 0.f;
  for (int k = threadIdx.x; k < LSTM_H; k += 64)
    s += hlast[(long)b * LSTM_H + k] * Wfc[k];
  red[threadIdx.x] = s;
  __syncthreads();
  for (int off = 32; off > 0; off >>= 1) {
    if (threadIdx.x < off) red[threadIdx.x] += red[threadIdx.x + off];
    __syncthreads();
  }
  if (threadIdx.x == 0) out[b] = 1.f / (1.f + __expf(-(red[0] + bfc[0])));
}

extern "C" void kernel_launch(void* const* d_in, const int* in_sizes, int n_in,
                              void* d_out, int out_size, void* d_ws, size_t ws_size,
                              hipStream_t stream)
{
  const float* x   = (const float*)d_in[0];
  const float* Wih = (const float*)d_in[1];
  const float* Whh = (const float*)d_in[2];
  const float* bih = (const float*)d_in[3];
  const float* bhh = (const float*)d_in[4];
  const float* Wfc = (const float*)d_in[5];
  const float* bfc = (const float*)d_in[6];
  char* ws = (char*)d_ws;

  lstm_prep<<<512, 256, 0, stream>>>(x, Wih, Whh, bih, bhh, ws);
  lstm_scan<<<NBLOCKS, 128, 0, stream>>>(ws);
  lstm_head<<<LSTM_B, 64, 0, stream>>>(ws, Wfc, bfc, (float*)d_out);
}
